// TransformerDecoder_33354716020981
// MI455X (gfx1250) — matrix operand order repaired and checked by reading, not compiled, not run on hardware
//
#include <hip/hip_runtime.h>
#include <math.h>

typedef __attribute__((ext_vector_type(16))) __bf16 v16bf;
typedef __attribute__((ext_vector_type(8)))  __bf16 v8bf;
typedef __attribute__((ext_vector_type(8)))  float  v8f;
typedef unsigned short u16b;

static constexpr int kE   = 768;
static constexpr int kS   = 512;
static constexpr int kB   = 4;
static constexpr int kBS  = 2048;   // B*S
static constexpr int kH   = 12;
static constexpr int kDK  = 64;
static constexpr int kHID = 3072;
static constexpr int kV   = 32000;
static constexpr int kL   = 4;

#define WMMA_BF16(a, b, c) \
  __builtin_amdgcn_wmma_f32_16x16x32_bf16(false, (a), false, (b), (short)0, (c), false, false)

// ---------- bf16 helpers ----------------------------------------------------
__device__ __forceinline__ u16b f2bf_bits(float f) {
  unsigned u = __builtin_bit_cast(unsigned, f);
  u += 0x7fffu + ((u >> 16) & 1u);
  return (u16b)(u >> 16);
}
__device__ __forceinline__ float gelu_f(float v) {
  return 0.5f * v * (1.0f + erff(v * 0.70710678118654752f));
}
// 16 contiguous bf16 (32B) -> v16bf  (two b128 loads)
__device__ __forceinline__ v16bf ld16(const u16b* p) {
  v8bf lo = *(const v8bf*)p;
  v8bf hi = *(const v8bf*)(p + 8);
  return __builtin_shufflevector(lo, hi, 0,1,2,3,4,5,6,7,8,9,10,11,12,13,14,15);
}
// two separate 8-element chunks -> v16bf
__device__ __forceinline__ v16bf ld2x8(const u16b* p0, const u16b* p1) {
  v8bf lo = *(const v8bf*)p0;
  v8bf hi = *(const v8bf*)p1;
  return __builtin_shufflevector(lo, hi, 0,1,2,3,4,5,6,7,8,9,10,11,12,13,14,15);
}

// ---------- positional vector ----------------------------------------------
__global__ void k_posvec(const float* __restrict__ pids,
                         const float* __restrict__ Wpos,
                         const float* __restrict__ bpos,
                         float* __restrict__ posv) {
  int e = blockIdx.x * 256 + threadIdx.x;
  float acc = bpos[e];
  for (int s = 0; s < kS; ++s) acc += pids[s] * Wpos[(size_t)s * kE + e];
  posv[e] = acc;
}

// ---------- embedding: one-hot gather + pos add ----------------------------
__global__ void k_embed(const float* __restrict__ X,
                        const float* __restrict__ Wemb,
                        const float* __restrict__ bemb,
                        const float* __restrict__ posv,
                        float* __restrict__ hf, u16b* __restrict__ hbf) {
  __shared__ int sidx;
  size_t row = blockIdx.x;
  if (threadIdx.x == 0) sidx = 0;
  __syncthreads();
  const float* xr = X + row * (size_t)kV;
  for (int i = threadIdx.x; i < kV; i += 256)
    if (xr[i] > 0.5f) sidx = i;          // one-hot: single writer
  __syncthreads();
  size_t idx = (size_t)sidx;
  for (int e = threadIdx.x; e < kE; e += 256) {
    float v = Wemb[idx * kE + e] + bemb[e] + posv[e];
    hf[row * kE + e]  = v;
    hbf[row * kE + e] = f2bf_bits(v);
  }
}

// ---------- weight converters (fragment-swizzled bf16) ----------------------
// Swizzled layout for B[K,N]: for tile (kt of 32 k, nt of 16 n):
//   dst[((kt*(N/16)+nt)*16 + n_local)*32 + kk] = B[kt*32+kk][nt*16+n_local]
__global__ void k_cvt_sw(const float* __restrict__ src, u16b* __restrict__ dst,
                         int N, int total) {
  int i = blockIdx.x * 256 + threadIdx.x;
  if (i >= total) return;
  int kk = i & 31;
  int nl = (i >> 5) & 15;
  int tile = i >> 9;
  int ntiles = N >> 4;
  int nt = tile % ntiles, kt = tile / ntiles;
  int k = kt * 32 + kk, n = nt * 16 + nl;
  dst[i] = f2bf_bits(src[(size_t)k * N + n]);
}
// [H,E,DK] weights -> swizzled B of logical [E, H*DK]
__global__ void k_cvt_qkv_sw(const float* __restrict__ src, u16b* __restrict__ dst) {
  int i = blockIdx.x * 256 + threadIdx.x;          // < H*E*DK
  int kk = i & 31;
  int nl = (i >> 5) & 15;
  int tile = i >> 9;
  const int ntiles = kE >> 4;                      // 48
  int nt = tile % ntiles, kt = tile / ntiles;
  int k = kt * 32 + kk;                            // = e
  int n = nt * 16 + nl;                            // = h*64 + dk
  int h = n >> 6, dk = n & 63;
  dst[i] = f2bf_bits(src[(size_t)h * (kE * kDK) + (size_t)k * kDK + dk]);
}

// ---------- WMMA bf16 GEMM (software-pipelined) -----------------------------
// C[M,N] = A[M,K](row-major bf16) @ Bsw[K,N](swizzled bf16) + bias, opt GELU.
// block = 256 = 8 waves (2m x 4n); wave tile 32m x 64n -> 8 WMMAs / k-step.
// swzout=1: write bf16 output in attention-V swizzle instead of row-major.
__global__ __launch_bounds__(256)
void k_gemm(const u16b* __restrict__ A, const u16b* __restrict__ Bm,
            const float* __restrict__ bias,
            float* __restrict__ outf, u16b* __restrict__ outbf,
            int N, int K, int act, int swzout) {
  int lane = threadIdx.x & 31, w = threadIdx.x >> 5;
  int wm = w & 1, wn = w >> 1;
  int m0 = blockIdx.y * 64 + wm * 32;
  int n0 = blockIdx.x * 256 + wn * 64;
  int mA = lane & 15, kbaseA = (lane < 16) ? 0 : 16;
  int nB = lane & 15, kbB    = (lane < 16) ? 0 : 16;
  int ntiles = N >> 4;
  int nt0 = n0 >> 4;

  const u16b* arow0 = A + (size_t)(m0 + mA) * K + kbaseA;
  const u16b* arow1 = arow0 + (size_t)16 * K;
  const u16b* bcol  = Bm + (((size_t)nt0) * 16 + nB) * 32 + kbB;
  const size_t bstep = (size_t)ntiles * 512;     // one k-tile of swizzled B

  v8f c[8] = {};
  // prologue: k-tile 0
  v16bf a0 = ld2x8(arow0, arow0 + 8);
  v16bf a1 = ld2x8(arow1, arow1 + 8);
  v16bf b0 = ld16(bcol);
  v16bf b1 = ld16(bcol + 512);
  v16bf b2 = ld16(bcol + 1024);
  v16bf b3 = ld16(bcol + 1536);

  for (int k0 = 32; k0 < K; k0 += 32) {
    // issue next k-tile's loads before consuming current fragments
    const u16b* ap0 = arow0 + k0;
    const u16b* ap1 = arow1 + k0;
    const u16b* bp  = bcol + (size_t)(k0 >> 5) * bstep;
    v16bf a0n = ld2x8(ap0, ap0 + 8);
    v16bf a1n = ld2x8(ap1, ap1 + 8);
    v16bf b0n = ld16(bp);
    v16bf b1n = ld16(bp + 512);
    v16bf b2n = ld16(bp + 1024);
    v16bf b3n = ld16(bp + 1536);

    c[0] = WMMA_BF16(a0, b0, c[0]);
    c[1] = WMMA_BF16(a0, b1, c[1]);
    c[2] = WMMA_BF16(a0, b2, c[2]);
    c[3] = WMMA_BF16(a0, b3, c[3]);
    c[4] = WMMA_BF16(a1, b0, c[4]);
    c[5] = WMMA_BF16(a1, b1, c[5]);
    c[6] = WMMA_BF16(a1, b2, c[6]);
    c[7] = WMMA_BF16(a1, b3, c[7]);

    a0 = a0n; a1 = a1n; b0 = b0n; b1 = b1n; b2 = b2n; b3 = b3n;
  }
  // tail k-tile
  c[0] = WMMA_BF16(a0, b0, c[0]);
  c[1] = WMMA_BF16(a0, b1, c[1]);
  c[2] = WMMA_BF16(a0, b2, c[2]);
  c[3] = WMMA_BF16(a0, b3, c[3]);
  c[4] = WMMA_BF16(a1, b0, c[4]);
  c[5] = WMMA_BF16(a1, b1, c[5]);
  c[6] = WMMA_BF16(a1, b2, c[6]);
  c[7] = WMMA_BF16(a1, b3, c[7]);

#pragma unroll
  for (int mi = 0; mi < 2; ++mi) {
#pragma unroll
    for (int nb = 0; nb < 4; ++nb) {
#pragma unroll
      for (int r = 0; r < 8; ++r) {
        int m = m0 + mi * 16 + ((lane < 16) ? r : 8 + r);
        int n = n0 + nb * 16 + (lane & 15);
        float v = c[mi * 4 + nb][r] + bias[n];
        if (act) v = gelu_f(v);
        if (outf) outf[(size_t)m * N + n] = v;
        if (outbf) {
          if (swzout) {
            // attention-V swizzle: (b,tt,h,nt,nn,kk)
            int b  = m >> 9, t = m & 511, tt = t >> 5, kk = t & 31;
            int h  = n >> 6, nt = (n >> 4) & 3, nn = n & 15;
            size_t o = ((((size_t)(b * 16 + tt) * kH + h) * 4 + nt) * 16 + nn) * 32 + kk;
            outbf[o] = f2bf_bits(v);
          } else {
            outbf[(size_t)m * N + n] = f2bf_bits(v);
          }
        }
      }
    }
  }
}

// ---------- fused causal flash attention per (b,h) -------------------------
// grid (S/64, H, B); block 128 = 4 waves; each wave owns a 16-row Q tile.
__global__ __launch_bounds__(128)
void k_attn(const u16b* __restrict__ Q, const u16b* __restrict__ Kb,
            const u16b* __restrict__ Vsw, u16b* __restrict__ O) {
  __shared__ alignas(16) u16b pbuf[4][16 * 32];
  int lane = threadIdx.x & 31, w = threadIdx.x >> 5;
  int b = blockIdx.z, h = blockIdx.y;
  int m0 = blockIdx.x * 64 + w * 16;
  int mA = lane & 15, kbaseA = (lane < 16) ? 0 : 16;
  int nB = lane & 15, kbB    = (lane < 16) ? 0 : 16;
  const float NEG = -3.0e38f;

  const u16b* qp = Q + (size_t)(b * kS + m0 + mA) * kE + h * kDK;
  v16bf aq0 = ld2x8(qp + kbaseA,      qp + 8 + kbaseA);
  v16bf aq1 = ld2x8(qp + 32 + kbaseA, qp + 40 + kbaseA);

  v8f o0 = {}, o1 = {}, o2 = {}, o3 = {};
  float mi[8], li[8];
#pragma unroll
  for (int r = 0; r < 8; ++r) { mi[r] = NEG; li[r] = 0.0f; }

  int tend = blockIdx.x * 64 + 63;          // uniform across the 4 waves
  for (int t0 = 0; t0 <= tend; t0 += 32) {
    // ---- scores: 4 WMMAs for two 16x16 tiles (Qtile @ Ktile^T) ----
    const u16b* kp = Kb + (size_t)(b * kS + t0 + nB) * kE + h * kDK;
    v16bf bk0 = ld16(kp + kbB);
    v16bf bk1 = ld16(kp + 32 + kbB);
    v16bf bk2 = ld16(kp + 16 * kE + kbB);
    v16bf bk3 = ld16(kp + 16 * kE + 32 + kbB);
    v8f s0 = {}, s1 = {};
    s0 = WMMA_BF16(aq0, bk0, s0);
    s0 = WMMA_BF16(aq1, bk1, s0);
    s1 = WMMA_BF16(aq0, bk2, s1);
    s1 = WMMA_BF16(aq1, bk3, s1);

    // ---- online softmax over the 32 new columns ----
    float alpha[8];
#pragma unroll
    for (int r = 0; r < 8; ++r) {
      int m  = m0 + ((lane < 16) ? r : 8 + r);
      int c0 = t0 + (lane & 15);
      int c1 = t0 + 16 + (lane & 15);
      float x0 = (c0 <= m) ? s0[r] * 0.125f : NEG;
      float x1 = (c1 <= m) ? s1[r] * 0.125f : NEG;
      float mx = fmaxf(x0, x1);
#pragma unroll
      for (int d = 1; d < 16; d <<= 1) mx = fmaxf(mx, __shfl_xor(mx, d, 32));
      float nm = fmaxf(mi[r], mx);
      float al = __expf(mi[r] - nm);
      float p0 = __expf(x0 - nm);
      float p1 = __expf(x1 - nm);
      float ps = p0 + p1;
#pragma unroll
      for (int d = 1; d < 16; d <<= 1) ps += __shfl_xor(ps, d, 32);
      li[r] = li[r] * al + ps;
      mi[r] = nm;
      alpha[r] = al;
      int ml = (lane < 16) ? r : 8 + r;
      pbuf[w][ml * 32 + (lane & 15)]      = f2bf_bits(p0);
      pbuf[w][ml * 32 + 16 + (lane & 15)] = f2bf_bits(p1);
    }
#pragma unroll
    for (int r = 0; r < 8; ++r) {
      o0[r] *= alpha[r]; o1[r] *= alpha[r]; o2[r] *= alpha[r]; o3[r] *= alpha[r];
    }
    __syncthreads();   // uniform trip count -> legal

    // ---- P tile (C-layout) -> A-fragment via LDS (b128 reads) ----
    const u16b* pr = &pbuf[w][(lane & 15) * 32];
    v16bf pa = ld2x8(pr + kbaseA, pr + 8 + kbaseA);

    // ---- V B-fragments from swizzled V: contiguous b128 loads ----
    int tt = t0 >> 5;
    const u16b* vb = Vsw + ((((size_t)(b * 16 + tt) * kH + h) * 4) * 16 + nB) * 32 + kbB;
    v16bf bv0 = ld16(vb);
    v16bf bv1 = ld16(vb + 512);
    v16bf bv2 = ld16(vb + 1024);
    v16bf bv3 = ld16(vb + 1536);
    o0 = WMMA_BF16(pa, bv0, o0);
    o1 = WMMA_BF16(pa, bv1, o1);
    o2 = WMMA_BF16(pa, bv2, o2);
    o3 = WMMA_BF16(pa, bv3, o3);
  }

#pragma unroll
  for (int r = 0; r < 8; ++r) {
    float inv = 1.0f / li[r];
    int m = m0 + ((lane < 16) ? r : 8 + r);
    size_t ro = (size_t)(b * kS + m) * kE + h * kDK + (lane & 15);
    O[ro]      = f2bf_bits(o0[r] * inv);
    O[ro + 16] = f2bf_bits(o1[r] * inv);
    O[ro + 32] = f2bf_bits(o2[r] * inv);
    O[ro + 48] = f2bf_bits(o3[r] * inv);
  }
}

// ---------- residual add + eps-free LayerNorm ------------------------------
__global__ void k_addln(const float* __restrict__ x, const float* __restrict__ d,
                        float* __restrict__ yf, u16b* __restrict__ ybf) {
  __shared__ float rowv[kE];
  __shared__ float red[256];
  size_t row = blockIdx.x;
  int tid = threadIdx.x;
  float p = 0.0f;
  for (int e = tid; e < kE; e += 256) {
    float v = x[row * kE + e] + d[row * kE + e];
    rowv[e] = v; p += v;
  }
  red[tid] = p; __syncthreads();
  for (int s = 128; s > 0; s >>= 1) { if (tid < s) red[tid] += red[tid + s]; __syncthreads(); }
  float mu = red[0] * (1.0f / kE);
  __syncthreads();
  float q = 0.0f;
  for (int e = tid; e < kE; e += 256) { float dd = rowv[e] - mu; q += dd * dd; }
  red[tid] = q; __syncthreads();
  for (int s = 128; s > 0; s >>= 1) { if (tid < s) red[tid] += red[tid + s]; __syncthreads(); }
  float rstd = rsqrtf(red[0] * (1.0f / kE));
  for (int e = tid; e < kE; e += 256) {
    float v = (rowv[e] - mu) * rstd;
    yf[row * kE + e]  = v;
    ybf[row * kE + e] = f2bf_bits(v);
  }
}

// ---------- final projection, last token only (fp32 bandwidth path) --------
__global__ void k_final(const float* __restrict__ hf, const float* __restrict__ Wout,
                        const float* __restrict__ bout, float* __restrict__ out) {
  __shared__ float hl[kB][kE];
  int tid = threadIdx.x;
  for (int i = tid; i < kB * kE; i += 256) {
    int bb = i / kE, e = i % kE;
    hl[bb][e] = hf[((size_t)bb * kS + (kS - 1)) * kE + e];
  }
  __syncthreads();
  int v = blockIdx.x * 256 + tid;
  float a0 = 0, a1 = 0, a2 = 0, a3 = 0;
  for (int e = 0; e < kE; ++e) {
    float wv = Wout[(size_t)e * kV + v];
    a0 += hl[0][e] * wv; a1 += hl[1][e] * wv;
    a2 += hl[2][e] * wv; a3 += hl[3][e] * wv;
  }
  float bb = bout[v];
  out[v]          = a0 + bb;
  out[kV + v]     = a1 + bb;
  out[2 * kV + v] = a2 + bb;
  out[3 * kV + v] = a3 + bb;
}

// ---------- host orchestration ---------------------------------------------
extern "C" void kernel_launch(void* const* d_in, const int* in_sizes, int n_in,
                              void* d_out, int out_size, void* d_ws, size_t ws_size,
                              hipStream_t stream) {
  const float* X    = (const float*)d_in[0];
  const float* pids = (const float*)d_in[1];
  const float* Wemb = (const float*)d_in[2];
  const float* bemb = (const float*)d_in[3];
  const float* Wpos = (const float*)d_in[4];
  const float* bpos = (const float*)d_in[5];
  const float* Wq   = (const float*)d_in[6];
  const float* bq   = (const float*)d_in[7];
  const float* Wk   = (const float*)d_in[8];
  const float* bk   = (const float*)d_in[9];
  const float* Wv   = (const float*)d_in[10];
  const float* bv   = (const float*)d_in[11];
  const float* Wo   = (const float*)d_in[12];
  const float* bo   = (const float*)d_in[13];
  const float* W1   = (const float*)d_in[14];
  const float* b1   = (const float*)d_in[15];
  const float* W2   = (const float*)d_in[16];
  const float* b2   = (const float*)d_in[17];
  const float* Wout = (const float*)d_in[18];
  const float* bout = (const float*)d_in[19];
  float* out = (float*)d_out;

  char* p = (char*)d_ws;
  auto carve = [&](size_t bytes) -> void* {
    void* r = (void*)p;
    p += (bytes + 255) & ~(size_t)255;
    return r;
  };
  float* posv  = (float*)carve(kE * 4);
  float* hf    = (float*)carve((size_t)kBS * kE * 4);
  u16b*  hbf   = (u16b*) carve((size_t)kBS * kE * 2);
  float* zf    = (float*)carve((size_t)kBS * kE * 4);
  u16b*  zbf   = (u16b*) carve((size_t)kBS * kE * 2);
  u16b*  qbf   = (u16b*) carve((size_t)kBS * kE * 2);
  u16b*  kbf   = (u16b*) carve((size_t)kBS * kE * 2);
  u16b*  vswz  = (u16b*) carve((size_t)kBS * kE * 2);
  u16b*  obf   = (u16b*) carve((size_t)kBS * kE * 2);
  float* accf  = (float*)carve((size_t)kBS * kE * 4);
  u16b*  ffbf  = (u16b*) carve((size_t)kBS * kHID * 2);
  u16b*  wqbf  = (u16b*) carve((size_t)kE * kE * 2);
  u16b*  wkbf  = (u16b*) carve((size_t)kE * kE * 2);
  u16b*  wvbf  = (u16b*) carve((size_t)kE * kE * 2);
  u16b*  wobf  = (u16b*) carve((size_t)kE * kE * 2);
  u16b*  w1bf  = (u16b*) carve((size_t)kE * kHID * 2);
  u16b*  w2bf  = (u16b*) carve((size_t)kHID * kE * 2);

  const int nQKV = kH * kE * kDK;      // 589824
  const int nFF  = kE * kHID;          // 2359296

  k_posvec<<<dim3(kE / 256), dim3(256), 0, stream>>>(pids, Wpos, bpos, posv);
  k_embed<<<dim3(kBS), dim3(256), 0, stream>>>(X, Wemb, bemb, posv, hf, hbf);

  dim3 gN768(kE / 256, kBS / 64);      // (3, 32)
  dim3 gN3072(kHID / 256, kBS / 64);   // (12, 32)

  for (int l = 0; l < kL; ++l) {
    // QKV projections (weights swizzled for fragment loads)
    k_cvt_qkv_sw<<<dim3(nQKV / 256), dim3(256), 0, stream>>>(Wq + (size_t)l * nQKV, wqbf);
    k_cvt_qkv_sw<<<dim3(nQKV / 256), dim3(256), 0, stream>>>(Wk + (size_t)l * nQKV, wkbf);
    k_cvt_qkv_sw<<<dim3(nQKV / 256), dim3(256), 0, stream>>>(Wv + (size_t)l * nQKV, wvbf);
    k_gemm<<<gN768, dim3(256), 0, stream>>>(hbf, wqbf, bq + (size_t)l * kE, nullptr, qbf, kE, kE, 0, 0);
    k_gemm<<<gN768, dim3(256), 0, stream>>>(hbf, wkbf, bk + (size_t)l * kE, nullptr, kbf, kE, kE, 0, 0);
    k_gemm<<<gN768, dim3(256), 0, stream>>>(hbf, wvbf, bv + (size_t)l * kE, nullptr, vswz, kE, kE, 0, 1);

    // fused causal attention
    k_attn<<<dim3(kS / 64, kH, kB), dim3(128), 0, stream>>>(qbf, kbf, vswz, obf);

    // output projection + residual + LN
    k_cvt_sw<<<dim3(nQKV / 256), dim3(256), 0, stream>>>(Wo + (size_t)l * nQKV, wobf, kE, nQKV);
    k_gemm<<<gN768, dim3(256), 0, stream>>>(obf, wobf, bo + (size_t)l * kE, accf, nullptr, kE, kE, 0, 0);
    k_addln<<<dim3(kBS), dim3(256), 0, stream>>>(hf, accf, zf, zbf);

    // FFN
    k_cvt_sw<<<dim3(nFF / 256), dim3(256), 0, stream>>>(W1 + (size_t)l * nFF, w1bf, kHID, nFF);
    k_gemm<<<gN3072, dim3(256), 0, stream>>>(zbf, w1bf, b1 + (size_t)l * kHID, nullptr, ffbf, kHID, kE, 1, 0);
    k_cvt_sw<<<dim3(nFF / 256), dim3(256), 0, stream>>>(W2 + (size_t)l * nFF, w2bf, kE, nFF);
    k_gemm<<<gN768, dim3(256), 0, stream>>>(ffbf, w2bf, b2 + (size_t)l * kE, accf, nullptr, kE, kHID, 0, 0);
    k_addln<<<dim3(kBS), dim3(256), 0, stream>>>(zf, accf, hf, hbf);
  }

  k_final<<<dim3(kV / 256), dim3(256), 0, stream>>>(hf, Wout, bout, out);
  (void)in_sizes; (void)n_in; (void)out_size; (void)ws_size;
}
